// MHDGcnNet_53206054863488
// MI455X (gfx1250) — matrix operand order repaired and checked by reading, not compiled, not run on hardware
//
#include <hip/hip_runtime.h>
#include <hip/hip_bf16.h>

// ---------------- CDNA5 WMMA helpers (wave32, 16x16x32 f16 -> f32) ----------

typedef __attribute__((ext_vector_type(16))) _Float16     v16h;
typedef __attribute__((ext_vector_type(8)))  float        v8f;
typedef __attribute__((ext_vector_type(4)))  unsigned int v4u;
typedef __attribute__((ext_vector_type(8)))  int          v8i;
typedef __attribute__((ext_vector_type(4)))  int          v4i;

__device__ __forceinline__ v8f wmma_f16(v16h a, v16h b, v8f c) {
    // D = A(16x32) * B(32x16) + C  — emits v_wmma_f32_16x16x32_f16
    return __builtin_amdgcn_wmma_f32_16x16x32_f16(
        /*neg_a=*/false, a, /*neg_b=*/false, b,
        /*c_mod=*/(short)0, c, /*reuse_a=*/false, /*reuse_b=*/false);
}

// A-operand gather from a row-major [16 x >=32] LDS tile (ISA 7.12.2):
__device__ __forceinline__ v16h ld_tile_A(const _Float16* t, int ldt, int lane) {
    int m = lane & 15, h = lane >> 4;
    v16h a;
#pragma unroll
    for (int v = 0; v < 8; ++v) {
        int kb = (v < 4) ? (h * 8 + 2 * v) : (16 + h * 8 + 2 * (v - 4));
        a[2 * v]     = t[m * ldt + kb];
        a[2 * v + 1] = t[m * ldt + kb + 1];
    }
    return a;
}

// B-operand gather from a row-major [32(K) x >=16(N)] LDS tile:
__device__ __forceinline__ v16h ld_tile_B(const _Float16* t, int ldt, int lane) {
    int n = lane & 15, h = lane >> 4;
    v16h b;
#pragma unroll
    for (int v = 0; v < 8; ++v) {
        int k = (v < 4) ? (h * 8 + 2 * v) : (16 + h * 8 + 2 * (v - 4));
        b[2 * v]     = t[k * ldt + n];
        b[2 * v + 1] = t[(k + 1) * ldt + n];
    }
    return b;
}

// C/D layout: VGPR r, lanes 0-15 -> M=r, N=lane ; lanes 16-31 -> M=8+r, N=lane-16
__device__ __forceinline__ void st_tile_f16(_Float16* out, int ldo, int lane, v8f c) {
    int n = lane & 15, h = lane >> 4;
#pragma unroll
    for (int r = 0; r < 8; ++r) out[(h * 8 + r) * ldo + n] = (_Float16)c[r];
}

// ---------------- CDNA5 Tensor Data Mover: 2-D f16 tile -> LDS --------------
// D# per ISA 08_async_tensor §8.3/8.4: group0 = {flags, lds_addr, global_addr,
// type=2}; group1 = {data_size=2B, tensor dims, tile dims, dim0 stride}.
// Issued once per wave; completion tracked with TENSORcnt.

#if defined(__clang_major__) && (__clang_major__ >= 23)
#define TDM_ISSUE(g0, g1, z4, z8) __builtin_amdgcn_tensor_load_to_lds(g0, g1, z4, z4, z8, 0)
#else
#define TDM_ISSUE(g0, g1, z4, z8) __builtin_amdgcn_tensor_load_to_lds(g0, g1, z4, z4, 0)
#endif

__device__ __forceinline__ void tdm_load_2d_f16(unsigned lds_off, const _Float16* gptr,
                                                unsigned dim0, unsigned dim1,
                                                unsigned stride0) {
    unsigned long long ga = (unsigned long long)(size_t)gptr;
    v4u g0;
    g0[0] = 1u;                                         // count=1 user descriptor
    g0[1] = lds_off;                                    // lds_addr (bytes)
    g0[2] = (unsigned)(ga & 0xffffffffu);               // global_addr[31:0]
    g0[3] = (unsigned)((ga >> 32) & 0x1ffffffu) | (2u << 30);  // addr[56:32] | type=2
    v8i g1;
    g1[0] = (int)(1u << 16);                            // data_size=1 -> 2 bytes/elem
    g1[1] = (int)(dim0 << 16);                          // tensor_dim0[15:0] @ bits 63:48
    g1[2] = (int)((dim0 >> 16) | (dim1 << 16));         // dim0[31:16] | dim1[15:0]
    g1[3] = (int)((dim1 >> 16) | ((dim0 & 0xffffu) << 16)); // dim1 hi | tile_dim0
    g1[4] = (int)(dim1 & 0xffffu);                      // tile_dim1 | tile_dim2=0
    g1[5] = (int)stride0;                               // tensor_dim0_stride[31:0]
    g1[6] = 0;
    g1[7] = 0;
    v4i z4 = {0, 0, 0, 0};
    v8i z8 = {0, 0, 0, 0, 0, 0, 0, 0};
    (void)z8;
    TDM_ISSUE(g0, g1, z4, z8);
}

// ---------------- constants --------------------------------------------------

#define BATCH 512
#define NNODE 64
#define CHAN  657

// reversed low-pass wavelet kernel (matches lax.conv with _LO[::-1])
__device__ __constant__ float LOREV[8] = {
    0.23037781330885523f,  0.7148465705525415f,  0.6308807679295904f,
   -0.02798376941698385f, -0.18703481171888114f, 0.030841381835986965f,
    0.032883011666982945f, -0.010597401784997278f };

// ---------------- Stage 1: channel avg/max over the 64 "width" positions ----

__global__ void k_cbam_stats(const float* __restrict__ x, float* __restrict__ avg,
                             float* __restrict__ mx) {
    int c = blockIdx.x * blockDim.x + threadIdx.x;
    int b = blockIdx.y;
    if (c >= CHAN) return;
    const float* xb = x + (size_t)b * NNODE * CHAN;
    float s = 0.f, m = -3.402823466e38f;
#pragma unroll 4
    for (int n = 0; n < NNODE; ++n) {
        float v = xb[n * CHAN + c];
        s += v; m = fmaxf(m, v);
    }
    avg[b * CHAN + c] = s * (1.f / 64.f);
    mx[b * CHAN + c]  = m;
}

// ---------------- Stage 2: MLP1 (WMMA): hsum = relu(avg@fc1^T)+relu(mx@fc1^T)

__global__ void k_mlp1(const float* __restrict__ avg, const float* __restrict__ mx,
                       const float* __restrict__ fc1, float* __restrict__ hsum) {
    __shared__ _Float16 sAa[16 * 32], sAm[16 * 32], sB[32 * 16];
    int lane = threadIdx.x;
    int m0 = blockIdx.x * 16, n0 = blockIdx.y * 16;
    v8f aa = {}, am = {};
    for (int k0 = 0; k0 < CHAN; k0 += 32) {
        for (int i = lane; i < 512; i += 32) {
            int m = i >> 5, k = i & 31, kg = k0 + k;
            float va = 0.f, vm = 0.f;
            if (kg < CHAN) { va = avg[(m0 + m) * CHAN + kg]; vm = mx[(m0 + m) * CHAN + kg]; }
            sAa[i] = (_Float16)va; sAm[i] = (_Float16)vm;
        }
        for (int i = lane; i < 512; i += 32) {
            int k = i >> 4, n = i & 15, kg = k0 + k, ng = n0 + n;
            sB[i] = (kg < CHAN && ng < 82) ? (_Float16)fc1[ng * CHAN + kg] : (_Float16)0.f;
        }
        __syncthreads();
        v16h bb = ld_tile_B(sB, 16, lane);
        aa = wmma_f16(ld_tile_A(sAa, 32, lane), bb, aa);
        am = wmma_f16(ld_tile_A(sAm, 32, lane), bb, am);
        __syncthreads();
    }
    int n = lane & 15, h = lane >> 4;
#pragma unroll
    for (int r = 0; r < 8; ++r)
        hsum[(m0 + h * 8 + r) * 96 + n0 + n] = fmaxf(aa[r], 0.f) + fmaxf(am[r], 0.f);
}

// ---------------- Stage 3: MLP2 (WMMA): ca = sigmoid(hsum @ fc2^T) ----------

__global__ void k_mlp2(const float* __restrict__ hsum, const float* __restrict__ fc2,
                       float* __restrict__ ca) {
    __shared__ _Float16 sA[16 * 32], sB[32 * 16];
    int lane = threadIdx.x;
    int m0 = blockIdx.x * 16, n0 = blockIdx.y * 16;
    v8f acc = {};
    for (int k0 = 0; k0 < 96; k0 += 32) {
        for (int i = lane; i < 512; i += 32) {
            int m = i >> 5, k = i & 31;
            sA[i] = (_Float16)hsum[(m0 + m) * 96 + k0 + k];
        }
        for (int i = lane; i < 512; i += 32) {
            int k = i >> 4, n = i & 15, kg = k0 + k, ng = n0 + n;
            sB[i] = (kg < 82 && ng < CHAN) ? (_Float16)fc2[ng * 82 + kg] : (_Float16)0.f;
        }
        __syncthreads();
        acc = wmma_f16(ld_tile_A(sA, 32, lane), ld_tile_B(sB, 16, lane), acc);
        __syncthreads();
    }
    int n = lane & 15, h = lane >> 4;
    if (n0 + n < CHAN) {
#pragma unroll
        for (int r = 0; r < 8; ++r)
            ca[(m0 + h * 8 + r) * CHAN + n0 + n] = 1.f / (1.f + expf(-acc[r]));
    }
}

// ---------------- Stage 4: spatial stats over channels (wave per (b,w)) -----

__global__ void k_spatial(const float* __restrict__ x, const float* __restrict__ ca,
                          float* __restrict__ smean, float* __restrict__ smax) {
    int lane = threadIdx.x;
    int idx = blockIdx.x * blockDim.y + threadIdx.y;   // b*64 + w
    int b = idx >> 6, w = idx & 63;
    const float* row = x + ((size_t)b * NNODE + w) * CHAN;
    const float* cab = ca + b * CHAN;
    float s = 0.f, m = -3.402823466e38f;
    for (int c = lane; c < CHAN; c += 32) {
        float v = row[c] * cab[c];
        s += v; m = fmaxf(m, v);
    }
#pragma unroll
    for (int o = 16; o > 0; o >>= 1) {
        s += __shfl_xor(s, o, 32);
        m = fmaxf(m, __shfl_xor(m, o, 32));
    }
    if (lane == 0) { smean[idx] = s * (1.f / 657.f); smax[idx] = m; }
}

// ---------------- Stage 5: 7-wide spatial conv + sigmoid --------------------

__global__ void k_saconv(const float* __restrict__ smean, const float* __restrict__ smax,
                         const float* __restrict__ saw, float* __restrict__ sa) {
    int b = blockIdx.x, w = threadIdx.x;
    float acc = 0.f;
#pragma unroll
    for (int j = 0; j < 7; ++j) {
        int ww = w + j - 3;
        if (ww >= 0 && ww < NNODE) {
            acc += saw[21 + j]      * smean[b * NNODE + ww];   // ch0, kernel row 3
            acc += saw[49 + 21 + j] * smax[b * NNODE + ww];    // ch1, kernel row 3
        }
    }
    sa[b * NNODE + w] = 1.f / (1.f + expf(-acc));
}

// ---------------- Stage 6: fused scale + 4-level wavelet cascade ------------
// Output delta is written directly in f16 (only consumed by WMMA GEMMs).

__device__ __forceinline__ void wstep(const float* in, int L, float* out, int Lo) {
    for (int t = threadIdx.x; t < Lo; t += blockDim.x) {
        float acc = 0.f;
#pragma unroll
        for (int k = 0; k < 8; ++k) {
            int i = 2 * t + k - 3;            // circular pad p=3
            if (i < 0) i += L;
            if (i >= L) i -= L;
            acc += LOREV[k] * in[i];
        }
        out[t] = acc;
    }
}

__global__ void k_wavelet(const float* __restrict__ x, const float* __restrict__ ca,
                          const float* __restrict__ sa, _Float16* __restrict__ delta) {
    __shared__ float bufA[CHAN];
    __shared__ float bufB[328];
    int idx = blockIdx.x;              // b*64 + n
    int b = idx >> 6, n = idx & 63;
    const float* row = x + ((size_t)b * NNODE + n) * CHAN;
    const float* cab = ca + b * CHAN;
    float sav = sa[b * NNODE + n];
    for (int c = threadIdx.x; c < CHAN; c += blockDim.x)
        bufA[c] = row[c] * cab[c] * sav;
    __syncthreads();
    wstep(bufA, 657, bufB, 328); __syncthreads();
    wstep(bufB, 328, bufA, 164); __syncthreads();
    wstep(bufA, 164, bufB, 82);  __syncthreads();
    _Float16* drow = delta + (size_t)idx * 48;
    for (int t = threadIdx.x; t < 41; t += blockDim.x) {
        float acc = 0.f;
#pragma unroll
        for (int k = 0; k < 8; ++k) {
            int i = 2 * t + k - 3;
            if (i < 0) i += 82;
            if (i >= 82) i -= 82;
            acc += LOREV[k] * bufB[i];
        }
        drow[t] = (_Float16)acc;
    }
    for (int t = 41 + threadIdx.x; t < 48; t += blockDim.x) drow[t] = (_Float16)0.f;
}

// ---------------- Stage 7: normalized adjacency (local + global) ------------

__global__ void k_adj(const float* __restrict__ lw, const float* __restrict__ tril,
                      const float* __restrict__ mask, float* __restrict__ Aout) {
    __shared__ float dinv[NNODE];
    __shared__ float W[NNODE * NNODE];
    int which = blockIdx.x;
    int r = threadIdx.x;
    float* A = Aout + which * NNODE * NNODE;
    float deg = 0.f;
    for (int c = 0; c < NNODE; ++c) {
        float w;
        if (which == 0) {
            w = lw[r * NNODE + c];
        } else {
            int i = (r >= c) ? (r * (r + 1) / 2 + c) : (c * (c + 1) / 2 + r);
            w = tril[i] * mask[r * NNODE + c];
        }
        W[r * NNODE + c] = w;
        deg += fabsf(w);
    }
    // all nodes have self-loop edges -> w_self == 0
    dinv[r] = (deg > 0.f) ? rsqrtf(deg) : 0.f;
    __syncthreads();
    for (int c = 0; c < NNODE; ++c)
        A[c * NNODE + r] = dinv[r] * W[r * NNODE + c] * dinv[c];   // A[col][row]
}

// ---------------- Stage 8: A -> A^5, emit f16 copy for the WMMA GEMMs -------

__global__ void k_apow(const float* __restrict__ Am, _Float16* __restrict__ A5h) {
    __shared__ float sA[4096], sP[4096], sT[4096];
    const float* A = Am + blockIdx.x * 4096;
    _Float16* Ah = A5h + blockIdx.x * 4096;
    int t = threadIdx.x;
    for (int i = t; i < 4096; i += 256) { sA[i] = A[i]; sP[i] = A[i]; }
    __syncthreads();
    for (int it = 0; it < 4; ++it) {
        for (int e = t; e < 4096; e += 256) {
            int i = e >> 6, j = e & 63;
            float acc = 0.f;
#pragma unroll 8
            for (int k = 0; k < 64; ++k) acc += sA[i * 64 + k] * sP[k * 64 + j];
            sT[e] = acc;
        }
        __syncthreads();
        for (int e = t; e < 4096; e += 256) sP[e] = sT[e];
        __syncthreads();
    }
    for (int i = t; i < 4096; i += 256) Ah[i] = (_Float16)sP[i];
}

// ---------------- Stage 9: GCN block 1 (TDM + WMMA) -------------------------
// z1 = (A^5 @ delta) @ W^T + b ; A^5 and delta tiles staged to LDS by the
// Tensor Data Mover (f16 in memory already), then consumed by WMMA.

__global__ void k_gcn1(const _Float16* __restrict__ A5h, const _Float16* __restrict__ delta,
                       const float* __restrict__ Wl, const float* __restrict__ bl,
                       const float* __restrict__ Wg, const float* __restrict__ bg,
                       float* __restrict__ z1) {
    __shared__ _Float16 sA[64 * 64];
    __shared__ _Float16 sX[64 * 48];
    __shared__ _Float16 sY[64 * 64];
    __shared__ _Float16 sW[64 * 16];
    int t = threadIdx.x, lane = t & 31, wid = t >> 5;
    int b = blockIdx.x, g = blockIdx.y;
    const float* W = g ? Wg : Wl;
    const float* bias = g ? bg : bl;
    if (t < 32) {   // wave 0 issues both tensor DMAs and waits on TENSORcnt
        tdm_load_2d_f16((unsigned)(size_t)&sA[0], A5h + g * 4096, 64, 64, 64);
        tdm_load_2d_f16((unsigned)(size_t)&sX[0], delta + (size_t)b * 64 * 48, 48, 64, 48);
        __builtin_amdgcn_s_wait_tensorcnt(0);
    }
    for (int i = t; i < 64 * 16; i += 128) {
        int k = i >> 4, n = i & 15;
        sW[i] = (k < 41) ? (_Float16)W[n * 41 + k] : (_Float16)0.f;
    }
    __syncthreads();
    int m0 = wid * 16;
    // Y(64x48) = A5(64x64) @ X(64x48)
    for (int nt = 0; nt < 3; ++nt) {
        v8f acc = {};
        for (int kc = 0; kc < 64; kc += 32) {
            v16h a  = ld_tile_A(&sA[m0 * 64 + kc], 64, lane);
            v16h bb = ld_tile_B(&sX[kc * 48 + nt * 16], 48, lane);
            acc = wmma_f16(a, bb, acc);
        }
        st_tile_f16(&sY[m0 * 64 + nt * 16], 64, lane, acc);
    }
    for (int i = lane; i < 256; i += 32) {     // zero K padding cols 48..63
        int r = i >> 4, c2 = i & 15;
        sY[(m0 + r) * 64 + 48 + c2] = (_Float16)0.f;
    }
    __syncthreads();
    // Z(64x16) = Y(64x48pad64) @ W^T(48pad64 x16) + b
    v8f acc = {};
    for (int kc = 0; kc < 64; kc += 32) {
        v16h a  = ld_tile_A(&sY[m0 * 64 + kc], 64, lane);
        v16h bb = ld_tile_B(&sW[kc * 16], 16, lane);
        acc = wmma_f16(a, bb, acc);
    }
    int n = lane & 15, h = lane >> 4;
    float bv = bias[n];
    float* zb = z1 + ((size_t)g * BATCH + b) * 64 * 16;
#pragma unroll
    for (int r = 0; r < 8; ++r) zb[(m0 + h * 8 + r) * 16 + n] = acc[r] + bv;
}

// ---------------- generic per-feature mean / rsqrt(var+eps) -----------------

__global__ void k_stats(const float* __restrict__ z, float* __restrict__ out,
                        int nfeat, long graphStride, int count) {
    int gb = blockIdx.x, g = gb / nfeat, f = gb % nfeat;
    const float* p = z + (size_t)g * graphStride + f;
    float s = 0.f, s2 = 0.f;
    for (int i = threadIdx.x; i < count; i += blockDim.x) {
        float v = p[(size_t)i * nfeat];
        s += v; s2 += v * v;
    }
    __shared__ float sh[256], sh2[256];
    int tid = threadIdx.x;
    sh[tid] = s; sh2[tid] = s2;
    __syncthreads();
    for (int o = 128; o > 0; o >>= 1) {
        if (tid < o) { sh[tid] += sh[tid + o]; sh2[tid] += sh2[tid + o]; }
        __syncthreads();
    }
    if (tid == 0) {
        float mean = sh[0] / count;
        float var  = sh2[0] / count - mean * mean;
        out[gb * 2]     = mean;
        out[gb * 2 + 1] = rsqrtf(var + 1e-5f);
    }
}

// ---------------- Stage 10: GCN block 2 (TDM + WMMA) ------------------------

__global__ void k_gcn2(const _Float16* __restrict__ A5h, const float* __restrict__ z1,
                       const float* __restrict__ st1,
                       const float* __restrict__ lg_, const float* __restrict__ lbe_,
                       const float* __restrict__ gg_, const float* __restrict__ gbe_,
                       const float* __restrict__ W1l, const float* __restrict__ b1l,
                       const float* __restrict__ W1g, const float* __restrict__ b1g,
                       float* __restrict__ z2) {
    __shared__ _Float16 sA[64 * 64];
    __shared__ _Float16 sX[64 * 16];
    __shared__ _Float16 sY[64 * 32];
    __shared__ _Float16 sW[32 * 16];
    int t = threadIdx.x, lane = t & 31, wid = t >> 5;
    int b = blockIdx.x, g = blockIdx.y;
    const float* gv  = g ? gg_ : lg_;
    const float* bev = g ? gbe_ : lbe_;
    const float* W   = g ? W1g : W1l;
    const float* bias= g ? b1g : b1l;
    if (t < 32) {   // wave 0 stages A^5 via the Tensor Data Mover
        tdm_load_2d_f16((unsigned)(size_t)&sA[0], A5h + g * 4096, 64, 64, 64);
        __builtin_amdgcn_s_wait_tensorcnt(0);
    }
    const float* zb = z1 + ((size_t)g * BATCH + b) * 64 * 16;
    const float* st = st1 + g * 32;
    for (int i = t; i < 64 * 16; i += 128) {
        int f = i & 15;
        float v = zb[i];
        v = (v - st[f * 2]) * st[f * 2 + 1] * gv[f] + bev[f];
        sX[i] = (_Float16)v;
    }
    for (int i = t; i < 32 * 16; i += 128) {
        int k = i >> 4, n = i & 15;
        sW[i] = (k < 16 && n < 4) ? (_Float16)W[n * 16 + k] : (_Float16)0.f;
    }
    __syncthreads();
    int m0 = wid * 16;
    v8f acc = {};
    for (int kc = 0; kc < 64; kc += 32) {
        v16h a  = ld_tile_A(&sA[m0 * 64 + kc], 64, lane);
        v16h bb = ld_tile_B(&sX[kc * 16], 16, lane);
        acc = wmma_f16(a, bb, acc);
    }
    st_tile_f16(&sY[m0 * 32], 32, lane, acc);
    for (int i = lane; i < 256; i += 32) {     // zero K padding cols 16..31
        int r = i >> 4, c2 = i & 15;
        sY[(m0 + r) * 32 + 16 + c2] = (_Float16)0.f;
    }
    __syncthreads();
    v8f acc2 = {};
    {
        v16h a  = ld_tile_A(&sY[m0 * 32], 32, lane);
        v16h bb = ld_tile_B(sW, 16, lane);
        acc2 = wmma_f16(a, bb, acc2);
    }
    int n = lane & 15, h = lane >> 4;
    if (n < 4) {
        float bv = bias[n];
        float* z2b = z2 + ((size_t)g * BATCH + b) * 64 * 4;
#pragma unroll
        for (int r = 0; r < 8; ++r) z2b[(m0 + h * 8 + r) * 4 + n] = acc2[r] + bv;
    }
}

// ---------------- Stage 11: normalize z2 + per-node feature means -----------

__global__ void k_feat(const float* __restrict__ z2, const float* __restrict__ st2,
                       const float* __restrict__ lg, const float* __restrict__ lbe,
                       const float* __restrict__ gg, const float* __restrict__ gbe,
                       float* __restrict__ feat) {
    int b = blockIdx.x, n = threadIdx.x;
    float acc[2];
#pragma unroll
    for (int g = 0; g < 2; ++g) {
        const float* gv = g ? gg : lg;
        const float* bv = g ? gbe : lbe;
        float s = 0.f;
#pragma unroll
        for (int c = 0; c < 4; ++c) {
            float v = z2[((size_t)g * BATCH + b) * 64 * 4 + n * 4 + c];
            s += (v - st2[(g * 4 + c) * 2]) * st2[(g * 4 + c) * 2 + 1] * gv[c] + bv[c];
        }
        acc[g] = s * 0.25f;
    }
    feat[b * 128 + n * 2 + 0] = acc[0];
    feat[b * 128 + n * 2 + 1] = acc[1];
}

// ---------------- Stage 12: classifier GEMM (WMMA) --------------------------

__global__ void k_cls1(const float* __restrict__ feat, const float* __restrict__ W1,
                       const float* __restrict__ b1, float* __restrict__ h1) {
    __shared__ _Float16 sA[16 * 32], sB[32 * 16];
    int lane = threadIdx.x;
    int m0 = blockIdx.x * 16, n0 = blockIdx.y * 16;
    v8f acc = {};
    for (int k0 = 0; k0 < 128; k0 += 32) {
        for (int i = lane; i < 512; i += 32) {
            int m = i >> 5, k = i & 31;
            sA[i] = (_Float16)feat[(m0 + m) * 128 + k0 + k];
        }
        for (int i = lane; i < 512; i += 32) {
            int k = i >> 4, n = i & 15;
            sB[i] = (_Float16)W1[(n0 + n) * 128 + k0 + k];
        }
        __syncthreads();
        acc = wmma_f16(ld_tile_A(sA, 32, lane), ld_tile_B(sB, 16, lane), acc);
        __syncthreads();
    }
    int n = lane & 15, h = lane >> 4;
    float bv = b1[n0 + n];
#pragma unroll
    for (int r = 0; r < 8; ++r) h1[(m0 + h * 8 + r) * 32 + n0 + n] = acc[r] + bv;
}

// ---------------- Stage 13: batchnorm + final linear + log_softmax ----------

__global__ void k_cls2(const float* __restrict__ h1, const float* __restrict__ st3,
                       const float* __restrict__ cg, const float* __restrict__ cbe,
                       const float* __restrict__ W2, const float* __restrict__ b2,
                       float* __restrict__ out) {
    int b = blockIdx.x * blockDim.x + threadIdx.x;
    if (b >= BATCH) return;
    float l0 = b2[0], l1 = b2[1];
#pragma unroll 8
    for (int f = 0; f < 32; ++f) {
        float xn = (h1[b * 32 + f] - st3[f * 2]) * st3[f * 2 + 1] * cg[f] + cbe[f];
        l0 += W2[f] * xn;
        l1 += W2[32 + f] * xn;
    }
    float mv  = fmaxf(l0, l1);
    float lse = mv + logf(expf(l0 - mv) + expf(l1 - mv));
    out[b * 2 + 0] = l0 - lse;
    out[b * 2 + 1] = l1 - lse;
}

// ---------------- launcher ---------------------------------------------------

extern "C" void kernel_launch(void* const* d_in, const int* in_sizes, int n_in,
                              void* d_out, int out_size, void* d_ws, size_t ws_size,
                              hipStream_t stream) {
    (void)in_sizes; (void)n_in; (void)out_size; (void)ws_size;
    const float* x     = (const float*)d_in[0];
    const float* lw    = (const float*)d_in[1];
    const float* gmask = (const float*)d_in[2];
    const float* fc1   = (const float*)d_in[3];
    const float* fc2   = (const float*)d_in[4];
    const float* saw   = (const float*)d_in[5];
    const float* tril  = (const float*)d_in[6];
    const float* ldW   = (const float*)d_in[7];
    const float* ldb   = (const float*)d_in[8];
    const float* ldg   = (const float*)d_in[9];
    const float* ldbe  = (const float*)d_in[10];
    const float* ld1W  = (const float*)d_in[11];
    const float* ld1b  = (const float*)d_in[12];
    const float* ld1g  = (const float*)d_in[13];
    const float* ld1be = (const float*)d_in[14];
    const float* dW    = (const float*)d_in[15];
    const float* db    = (const float*)d_in[16];
    const float* dg    = (const float*)d_in[17];
    const float* dbe   = (const float*)d_in[18];
    const float* d1W   = (const float*)d_in[19];
    const float* d1b   = (const float*)d_in[20];
    const float* d1g   = (const float*)d_in[21];
    const float* d1be  = (const float*)d_in[22];
    const float* cW1   = (const float*)d_in[23];
    const float* cb1   = (const float*)d_in[24];
    const float* cg    = (const float*)d_in[25];
    const float* cbe   = (const float*)d_in[26];
    const float* cW2   = (const float*)d_in[27];
    const float* cb2   = (const float*)d_in[28];
    // d_in[29] = edges (full 64x64 graph; handled analytically)

    float* ws = (float*)d_ws;
    float* AVG   = ws;
    float* MX    = AVG   + (size_t)BATCH * CHAN;
    float* HSUM  = MX    + (size_t)BATCH * CHAN;
    float* CA    = HSUM  + (size_t)BATCH * 96;
    float* SMEAN = CA    + (size_t)BATCH * CHAN;
    float* SMAX  = SMEAN + (size_t)BATCH * NNODE;
    float* SA    = SMAX  + (size_t)BATCH * NNODE;
    float* DELF  = SA    + (size_t)BATCH * NNODE;          // f16 delta region
    _Float16* DELTAH = (_Float16*)DELF;                    // BATCH*64*48 halves
    float* AMAT  = DELF  + (size_t)BATCH * NNODE * 48 / 2; // f32 A (pre-power)
    float* A5F   = AMAT  + 2 * 4096;                       // f16 A^5 region
    _Float16* A5H = (_Float16*)A5F;                        // 2*4096 halves
    float* Z1    = A5F   + 4096;
    float* ST1   = Z1    + (size_t)2 * BATCH * NNODE * 16;
    float* Z2    = ST1   + 64;
    float* ST2   = Z2    + (size_t)2 * BATCH * NNODE * 4;
    float* FEAT  = ST2   + 16;
    float* H1    = FEAT  + (size_t)BATCH * 128;
    float* ST3   = H1    + (size_t)BATCH * 32;

    k_cbam_stats<<<dim3(3, BATCH), 256, 0, stream>>>(x, AVG, MX);
    k_mlp1<<<dim3(32, 6), 32, 0, stream>>>(AVG, MX, fc1, HSUM);
    k_mlp2<<<dim3(32, 42), 32, 0, stream>>>(HSUM, fc2, CA);
    k_spatial<<<dim3(BATCH * NNODE / 8), dim3(32, 8), 0, stream>>>(x, CA, SMEAN, SMAX);
    k_saconv<<<BATCH, NNODE, 0, stream>>>(SMEAN, SMAX, saw, SA);
    k_wavelet<<<BATCH * NNODE, 128, 0, stream>>>(x, CA, SA, DELTAH);
    k_adj<<<2, NNODE, 0, stream>>>(lw, tril, gmask, AMAT);
    k_apow<<<2, 256, 0, stream>>>(AMAT, A5H);
    k_gcn1<<<dim3(BATCH, 2), 128, 0, stream>>>(A5H, DELTAH, ldW, ldb, dW, db, Z1);
    k_stats<<<32, 256, 0, stream>>>(Z1, ST1, 16, (long)BATCH * NNODE * 16, BATCH * NNODE);
    k_gcn2<<<dim3(BATCH, 2), 128, 0, stream>>>(A5H, Z1, ST1, ldg, ldbe, dg, dbe,
                                               ld1W, ld1b, d1W, d1b, Z2);
    k_stats<<<8, 256, 0, stream>>>(Z2, ST2, 4, (long)BATCH * NNODE * 4, BATCH * NNODE);
    k_feat<<<BATCH, NNODE, 0, stream>>>(Z2, ST2, ld1g, ld1be, d1g, d1be, FEAT);
    k_cls1<<<dim3(32, 2), 32, 0, stream>>>(FEAT, cW1, cb1, H1);
    k_stats<<<32, 256, 0, stream>>>(H1, ST3, 32, 0L, BATCH);
    k_cls2<<<2, 256, 0, stream>>>(H1, ST3, cg, cbe, cW2, cb2, (float*)d_out);
}
